// DeformConvWithOffsetScaleGaussBoundMinMaxShared_28647431864983
// MI455X (gfx1250) — compile-verified, matrix operand order checked by reading, not run on hardware
//
#include <hip/hip_runtime.h>

// ---------------------------------------------------------------------------
// Deformable conv 3x3 (two offset branches summed) for MI455X / gfx1250.
// Implicit GEMM: M = N*H*W = 32768 pixels, Ndim = 128 out channels,
// K = 18 kernel points * 128 in channels = 2304, via v_wmma_f32_16x16x32_bf16.
// Block tile: 32 pixels x 128 out channels; 8 waves, 2 accumulators each.
// Double-buffered LDS A (1 barrier per K-step), packed bf16 dword stores.
// ---------------------------------------------------------------------------

#define Nn   8
#define Cc   128
#define Oo   128
#define Hh   64
#define Ww   64
#define HW   (Hh * Ww)          // 4096
#define KK   18                 // 9 points x 2 branches
#define KRED (KK * Cc)          // 2304
#define KSTEPS (KRED / 32)      // 72

typedef __attribute__((ext_vector_type(16))) __bf16 v16bf;
typedef __attribute__((ext_vector_type(8)))  float  v8f;

// 9 kernel points, (y,x) pairs row-major
__constant__ float AD[18] = {-1.f,-1.f, -1.f,0.f, -1.f,1.f,
                              0.f,-1.f,  0.f,0.f,  0.f,1.f,
                              1.f,-1.f,  1.f,0.f,  1.f,1.f};
__constant__ float AG[18] = {-0.7071f,-0.7071f, -1.f,0.f, -0.7071f,0.7071f,
                              0.f,-1.f,          0.f,0.f,  0.f,1.f,
                              0.7071f,-0.7071f,  1.f,0.f,  0.7071f,0.7071f};

__device__ __forceinline__ unsigned short f32_to_bf16(float f) {
    union { float f; unsigned u; } v; v.f = f;
    unsigned u = v.u;
    u += 0x7FFFu + ((u >> 16) & 1u);      // round to nearest even
    return (unsigned short)(u >> 16);
}

__device__ __forceinline__ unsigned pack_bf16x2(float lo, float hi) {
    return (unsigned)f32_to_bf16(lo) | ((unsigned)f32_to_bf16(hi) << 16);
}

// map k_local (0..31) within a 16x32 bf16 A/B tile to (lane-half h, vgpr j)
__device__ __forceinline__ void klocal_map(int kl, int& h, int& j) {
    if (kl < 16) { h = kl >> 3; j = (kl & 7) >> 1; }
    else         { int k2 = kl - 16; h = k2 >> 3; j = 4 + ((k2 & 7) >> 1); }
}

// ---------------------------------------------------------------------------
// Kernel 1: weights -> bf16 B fragments, pre-swizzled to per-lane order.
// Bws dword id = ((ks*8 + og)*32 + lane)*8 + j  holds K pair (k0,k0+1).
// ---------------------------------------------------------------------------
__global__ __launch_bounds__(256) void wprep_kernel(
    const float* __restrict__ weight, unsigned* __restrict__ Bws)
{
    int id = blockIdx.x * 256 + threadIdx.x;
    if (id >= KSTEPS * 8 * 32 * 8) return;
    int j    = id & 7;
    int lane = (id >> 3) & 31;
    int og   = (id >> 8) & 7;
    int ks   = id >> 11;
    int h    = lane >> 4;
    int o    = og * 16 + (lane & 15);
    int k0   = (j < 4) ? (8 * h + 2 * j) : (16 + 8 * h + 2 * (j - 4));
    int kk   = ks >> 2;          // kernel point 0..17
    int kk9  = (kk >= 9) ? kk - 9 : kk;
    int c0   = ((ks & 3) << 5) + k0;
    Bws[id] = pack_bf16x2(weight[(o * Cc + c0    ) * 9 + kk9],
                          weight[(o * Cc + c0 + 1) * 9 + kk9]);
}

// ---------------------------------------------------------------------------
// Kernel 2: 2-channel scale conv + clamp + per-point sample coords.
// ---------------------------------------------------------------------------
__global__ __launch_bounds__(256) void scale_offset_kernel(
    const float* __restrict__ x, const float* __restrict__ scale_w,
    const float* __restrict__ scale_b,
    float* __restrict__ pyA, float* __restrict__ pxA)
{
    __shared__ float sw[2 * Cc * 9];                 // 9.2 KB
    for (int i = threadIdx.x; i < 2 * Cc * 9; i += 256) sw[i] = scale_w[i];
    __syncthreads();

    int g  = blockIdx.x * 256 + threadIdx.x;         // 0..32767
    int n  = g >> 12;
    int y  = (g >> 6) & 63;
    int xg = g & 63;

    const float* xn = x + (size_t)n * Cc * HW;
    float a0 = scale_b[0], a1 = scale_b[1];
    for (int c = 0; c < Cc; ++c) {
        const float* xc = xn + c * HW;
        const float* w0 = &sw[c * 9];
        const float* w1 = &sw[Cc * 9 + c * 9];
#pragma unroll
        for (int t = 0; t < 9; ++t) {
            int yy = y + t / 3 - 1, xx = xg + t % 3 - 1;
            float v = (yy >= 0 && yy < Hh && xx >= 0 && xx < Ww)
                          ? xc[yy * Ww + xx] : 0.f;
            a0 = fmaf(v, w0[t], a0);
            a1 = fmaf(v, w1[t], a1);
        }
    }
    float s0 = fminf(fmaxf(a0, 0.f), 8.f);     // Hardtanh(0, bound_min)
    float s1 = fminf(fmaxf(a1, 8.f), 16.f);    // Hardtanh(bound_min, bound_max)

#pragma unroll
    for (int kk = 0; kk < KK; ++kk) {
        int   k9 = (kk >= 9) ? kk - 9 : kk;
        float s  = (kk < 9) ? s0 : s1;
        float dy = s * AG[2 * k9]     - AD[2 * k9];
        float dx = s * AG[2 * k9 + 1] - AD[2 * k9 + 1];
        float py = (float)(y  - 1 + k9 / 3) + dy;
        float px = (float)(xg - 1 + k9 % 3) + dx;
        int idx = ((n * KK + kk) * Hh + y) * Ww + xg;
        pyA[idx] = py;
        pxA[idx] = px;
    }
}

// ---------------------------------------------------------------------------
// Kernel 3: fused im2col gather + WMMA GEMM.
// One block = 32 consecutive pixels (n, y, xb..xb+31) x all 128 out channels.
// 8 waves; wave w owns output channels [16w, 16w+16), two 16-pixel subtiles.
// ---------------------------------------------------------------------------
struct Samp { float w0, w1, w2, w3; int i0, i1, i2, i3; };

__global__ __launch_bounds__(256) void deform_gemm_kernel(
    const float* __restrict__ x, const float* __restrict__ bias,
    const unsigned* __restrict__ Bws,
    const float* __restrict__ pyA, const float* __restrict__ pxA,
    float* __restrict__ out)
{
    __shared__ Samp samp[32 * KK];                          // 18 KB
    __shared__ __align__(32) unsigned ldsA[2][2][32][8];    // dbl-buf, 2 subtiles

    int bid = blockIdx.x;                // 1024 blocks
    int n   = bid >> 7;                  // 128 tiles per image (64 rows x 2)
    int rem = bid & 127;
    int y   = rem >> 1;
    int xb  = (rem & 1) << 5;

    int tid  = threadIdx.x;
    int lane = tid & 31;
    int wave = tid >> 5;

    // -- precompute bilinear corner weights / indices per (pixel, point kk)
    for (int e = tid; e < 32 * KK; e += 256) {
        int m  = e & 31;
        int kk = e >> 5;
        int pidx = ((n * KK + kk) * Hh + y) * Ww + xb + m;
        float py = pyA[pidx], px = pxA[pidx];
        float fy = floorf(py), fx = floorf(px);
        int   y0 = (int)fy,   x0 = (int)fx;
        float wy = py - fy,   wx = px - fx;
        Samp s;
        {
            int yi = y0, xi = x0; float wt = (1.f - wy) * (1.f - wx);
            bool ok = yi >= 0 && yi < Hh && xi >= 0 && xi < Ww;
            s.w0 = ok ? wt : 0.f;
            s.i0 = min(max(yi, 0), Hh - 1) * Ww + min(max(xi, 0), Ww - 1);
        }
        {
            int yi = y0, xi = x0 + 1; float wt = (1.f - wy) * wx;
            bool ok = yi >= 0 && yi < Hh && xi >= 0 && xi < Ww;
            s.w1 = ok ? wt : 0.f;
            s.i1 = min(max(yi, 0), Hh - 1) * Ww + min(max(xi, 0), Ww - 1);
        }
        {
            int yi = y0 + 1, xi = x0; float wt = wy * (1.f - wx);
            bool ok = yi >= 0 && yi < Hh && xi >= 0 && xi < Ww;
            s.w2 = ok ? wt : 0.f;
            s.i2 = min(max(yi, 0), Hh - 1) * Ww + min(max(xi, 0), Ww - 1);
        }
        {
            int yi = y0 + 1, xi = x0 + 1; float wt = wy * wx;
            bool ok = yi >= 0 && yi < Hh && xi >= 0 && xi < Ww;
            s.w3 = ok ? wt : 0.f;
            s.i3 = min(max(yi, 0), Hh - 1) * Ww + min(max(xi, 0), Ww - 1);
        }
        samp[e] = s;
    }
    __syncthreads();

    const float* xn = x + (size_t)n * Cc * HW;
    const v16bf* Bf = reinterpret_cast<const v16bf*>(Bws);

    // gather-thread invariants: pixel m, consecutive channel pair (2pr, 2pr+1)
    int m  = tid & 15;
    int pr = tid >> 4;                 // 0..15
    int cl0 = 2 * pr;
    int hA, jA;
    klocal_map(cl0, hA, jA);
    int rowA = m + 16 * hA;

    v8f acc0 = {}, acc1 = {};
    for (int kk = 0; kk < KK; ++kk) {
        // sample params for this kernel point: subtile 0 pixel m, subtile 1 pixel m+16
        Samp sA = samp[kk * 32 + m];
        Samp sB = samp[kk * 32 + 16 + m];
#pragma unroll
        for (int ch = 0; ch < 4; ++ch) {
            int ks  = kk * 4 + ch;
            int buf = ch & 1;
            // B fragment (global, independent of LDS) — issue first
            v16bf bfrag = Bf[(ks * 8 + wave) * 32 + lane];

            const float* xp0 = xn + (size_t)((ch << 5) + cl0) * HW;
            const float* xp1 = xp0 + HW;
            // subtile 0, channels cl0 / cl0+1
            float a0 = fmaf(sA.w0, xp0[sA.i0], fmaf(sA.w1, xp0[sA.i1],
                       fmaf(sA.w2, xp0[sA.i2], sA.w3 * xp0[sA.i3])));
            float a1 = fmaf(sA.w0, xp1[sA.i0], fmaf(sA.w1, xp1[sA.i1],
                       fmaf(sA.w2, xp1[sA.i2], sA.w3 * xp1[sA.i3])));
            // subtile 1
            float b0 = fmaf(sB.w0, xp0[sB.i0], fmaf(sB.w1, xp0[sB.i1],
                       fmaf(sB.w2, xp0[sB.i2], sB.w3 * xp0[sB.i3])));
            float b1 = fmaf(sB.w0, xp1[sB.i0], fmaf(sB.w1, xp1[sB.i1],
                       fmaf(sB.w2, xp1[sB.i2], sB.w3 * xp1[sB.i3])));

            ldsA[buf][0][rowA][jA] = pack_bf16x2(a0, a1);
            ldsA[buf][1][rowA][jA] = pack_bf16x2(b0, b1);
            __syncthreads();

            v16bf af0 = *reinterpret_cast<const v16bf*>(&ldsA[buf][0][lane][0]);
            v16bf af1 = *reinterpret_cast<const v16bf*>(&ldsA[buf][1][lane][0]);
            acc0 = __builtin_amdgcn_wmma_f32_16x16x32_bf16(
                       false, af0, false, bfrag, (short)0, acc0, false, false);
            acc1 = __builtin_amdgcn_wmma_f32_16x16x32_bf16(
                       false, af1, false, bfrag, (short)0, acc1, false, false);
            // no trailing barrier: next step writes the other LDS buffer
        }
    }

    // epilogue: bias added twice in the reference (once per branch)
    int o = wave * 16 + (lane & 15);
    float b2 = 2.0f * bias[o];
    float* op = out + ((size_t)(n * Oo + o) * Hh + y) * Ww + xb + 8 * (lane >> 4);
#pragma unroll
    for (int v = 0; v < 8; ++v) op[v]      = acc0[v] + b2;
#pragma unroll
    for (int v = 0; v < 8; ++v) op[16 + v] = acc1[v] + b2;
}

// ---------------------------------------------------------------------------
extern "C" void kernel_launch(void* const* d_in, const int* in_sizes, int n_in,
                              void* d_out, int out_size, void* d_ws, size_t ws_size,
                              hipStream_t stream) {
    const float* x       = (const float*)d_in[0];
    const float* weight  = (const float*)d_in[1];
    const float* bias    = (const float*)d_in[2];
    const float* scale_w = (const float*)d_in[3];
    const float* scale_b = (const float*)d_in[4];
    float* out = (float*)d_out;

    // workspace layout (all 256B aligned):
    //   Bws : 72*8*32*8 dwords = 589824 B
    //   pyA : 8*18*4096 floats = 2359296 B
    //   pxA : 8*18*4096 floats = 2359296 B
    char* ws = (char*)d_ws;
    unsigned* Bws = (unsigned*)ws;
    float* pyA = (float*)(ws + 589824);
    float* pxA = (float*)(ws + 589824 + 2359296);

    wprep_kernel<<<(KSTEPS * 8 * 32 * 8 + 255) / 256, 256, 0, stream>>>(weight, Bws);
    scale_offset_kernel<<<(Nn * HW) / 256, 256, 0, stream>>>(x, scale_w, scale_b, pyA, pxA);
    deform_gemm_kernel<<<Nn * (HW / 32), 256, 0, stream>>>(x, bias, Bws, pyA, pxA, out);
}
